// WTNet_87479893885470
// MI455X (gfx1250) — matrix operand order repaired and checked by reading, not compiled, not run on hardware
//
#include <hip/hip_runtime.h>

// ---- problem constants (match reference) ----
#define B_      4
#define NNEG_   33
#define N_      10000
#define E_      100000
#define W_      3
#define D_      64
#define L_      3
#define R_      200
#define FLAYERS_ 2
#define FFN_    256
#define EPS_    1e-5f

typedef __attribute__((ext_vector_type(16))) _Float16 v16h;
typedef __attribute__((ext_vector_type(8)))  float    v8f;

// ----------------------------------------------------------------------------
// prep: min_t over edge_time[0]; is_t_neg per batch; h0 / tsel / query gather
// ----------------------------------------------------------------------------
__global__ void wt_prep_kernel(const int* __restrict__ qt,        // (B,NNEG,3)
                               const float* __restrict__ qemb,    // (R,D)
                               const int* __restrict__ etime0,    // (E,) window 0
                               int* __restrict__ h0,              // (B,)
                               int* __restrict__ tsel,            // (B,NNEG)
                               float* __restrict__ query,         // (B,D)
                               int* __restrict__ mint)            // (1,)
{
    __shared__ int sh_min[256];
    __shared__ int sh_neg[B_];
    int tid = threadIdx.x;

    int m = 0x7FFFFFFF;
    for (int i = tid; i < E_; i += 256) m = min(m, etime0[i]);
    sh_min[tid] = m;
    __syncthreads();
    for (int s = 128; s > 0; s >>= 1) {
        if (tid < s) sh_min[tid] = min(sh_min[tid], sh_min[tid + s]);
        __syncthreads();
    }
    if (tid == 0) *mint = sh_min[0];

    if (tid < B_) {
        const int* row = qt + tid * NNEG_ * 3;
        int hh = row[0];
        int neg = 1;
        for (int j = 1; j < NNEG_; ++j)
            if (row[j * 3 + 0] != hh) { neg = 0; break; }
        sh_neg[tid] = neg;
        h0[tid] = neg ? row[0] : row[2];
    }
    __syncthreads();

    for (int idx = tid; idx < B_ * NNEG_; idx += 256) {
        int b = idx / NNEG_, j = idx % NNEG_;
        const int* tr = qt + (b * NNEG_ + j) * 3;
        tsel[idx] = sh_neg[b] ? tr[2] : tr[0];
    }
    for (int idx = tid; idx < B_ * D_; idx += 256) {
        int b = idx >> 6, d = idx & 63;
        const int* row = qt + b * NNEG_ * 3;
        int r0 = sh_neg[b] ? row[1] : row[1] + (R_ / 2);
        query[idx] = qemb[r0 * D_ + d];
    }
}

// ----------------------------------------------------------------------------
// convert lin_w (L,D,2D) f32 -> f16 workspace copy
// ----------------------------------------------------------------------------
__global__ void wt_convw16_kernel(const float* __restrict__ lw, _Float16* __restrict__ w16)
{
    int idx = blockIdx.x * blockDim.x + threadIdx.x;
    if (idx < L_ * D_ * 2 * D_) w16[idx] = (_Float16)lw[idx];
}

// ----------------------------------------------------------------------------
// rel_all[(l*B+b)*R*D + i] = sum_d query[b][d] * rel_lin_w[l][i][d] + rel_lin_b[l][i]
// ----------------------------------------------------------------------------
__global__ void wt_rel_kernel(const float* __restrict__ query,
                              const float* __restrict__ rlw,   // (L, R*D, D)
                              const float* __restrict__ rlb,   // (L, R*D)
                              float* __restrict__ relall)      // (L, B, R*D)
{
    int gid = blockIdx.x * blockDim.x + threadIdx.x;
    const int RD = R_ * D_;
    if (gid >= L_ * B_ * RD) return;
    int i = gid % RD;
    int b = (gid / RD) % B_;
    int l = gid / (RD * B_);
    const float* q    = query + b * D_;
    const float* wrow = rlw + ((size_t)l * RD + i) * D_;
    float acc = rlb[(size_t)l * RD + i];
#pragma unroll 8
    for (int d = 0; d < D_; ++d) acc += q[d] * wrow[d];
    relall[gid] = acc;
}

// ----------------------------------------------------------------------------
// buf[b,n,:] = (n == h0[b]) ? query[b,:] : 0        (== "init" in reference)
// ----------------------------------------------------------------------------
__global__ void wt_init_state_kernel(float* __restrict__ buf,
                                     const float* __restrict__ query,
                                     const int* __restrict__ h0)
{
    int idx = blockIdx.x * blockDim.x + threadIdx.x;
    if (idx >= B_ * N_ * D_) return;
    int d = idx & 63;
    int n = (idx >> 6) % N_;
    int b = idx / (N_ * D_);
    buf[idx] = (n == h0[b]) ? query[b * D_ + d] : 0.f;
}

// ----------------------------------------------------------------------------
// edge scatter: one wave per edge, 2 dims per lane, 4 batches
// agg[b,dst,d] += x[b,src,d] * rel[b,etype,d] * cos(dt*tw[etype,d]+tb[etype,d]) * ew
// ----------------------------------------------------------------------------
__global__ void wt_edge_kernel(const int* __restrict__ src,
                               const int* __restrict__ dst,
                               const int* __restrict__ etype,
                               const int* __restrict__ etime,
                               const float* __restrict__ ew,
                               const float* __restrict__ x,      // (B,N,D)
                               const float* __restrict__ rel,    // (B,R,D) for layer l
                               const float* __restrict__ tw,     // (R,D) layer l
                               const float* __restrict__ tb,     // (R,D) layer l
                               const int* __restrict__ mint,
                               float* __restrict__ agg)          // (B,N,D)
{
    int wid  = (blockIdx.x * blockDim.x + threadIdx.x) >> 5;
    int lane = threadIdx.x & 31;
    if (wid >= E_) return;
    int s  = src[wid];
    int dd = dst[wid];
    int et = etype[wid];
    float dt = (float)(etime[wid] - *mint);
    float w  = ew[wid];
    int d0 = lane, d1 = lane + 32;
    const float* twr = tw + (size_t)et * D_;
    const float* tbr = tb + (size_t)et * D_;
    float te0 = __cosf(dt * twr[d0] + tbr[d0]) * w;   // fast cos; args are O(1..100)*O(0.05)
    float te1 = __cosf(dt * twr[d1] + tbr[d1]) * w;
#pragma unroll
    for (int b = 0; b < B_; ++b) {
        const float* xr = x   + ((size_t)b * N_ + s)  * D_;
        const float* rr = rel + ((size_t)b * R_ + et) * D_;
        float*       ar = agg + ((size_t)b * N_ + dd) * D_;
        atomicAdd(&ar[d0], xr[d0] * rr[d0] * te0);
        atomicAdd(&ar[d1], xr[d1] * rr[d1] * te1);
    }
}

// ----------------------------------------------------------------------------
// fused GEMM + bias + LayerNorm + ReLU + residual, WMMA f32<-f16 16x16x32.
// block = 128 threads (4 waves) handles 16 nodes of one batch.
//   hid[m,o] = sum_k cat[m,k] * lin_w[o,k],  cat = [x | agg], K=128, o<64
//   x[m,o]  = relu(LN(hid)[m,o]) + x[m,o]
// ----------------------------------------------------------------------------
__global__ void wt_gemm_ln_kernel(float* __restrict__ x,          // (B,N,D) in/out
                                  const float* __restrict__ agg,  // (B,N,D)
                                  const _Float16* __restrict__ w16, // (D, 2D) layer l
                                  const float* __restrict__ bias, // (D,)
                                  const float* __restrict__ lns,  // (D,)
                                  const float* __restrict__ lnb)  // (D,)
{
    __shared__ _Float16 catl[16][2 * D_];   // 16x128 f16, 4KB
    __shared__ float    hidl[16][D_];       // 16x64 f32, 4KB
    __shared__ float    red[16][8][2];

    int blk   = blockIdx.x;
    int b     = blk / (N_ / 16);
    int node0 = (blk % (N_ / 16)) * 16;
    int tid   = threadIdx.x;

    const float* xb = x   + ((size_t)b * N_) * D_;
    const float* ab = agg + ((size_t)b * N_) * D_;

    // stage concat(x, agg) -> LDS f16
    for (int idx = tid; idx < 16 * 2 * D_; idx += 128) {
        int row = idx >> 7, k = idx & 127;
        float v = (k < D_) ? xb[(size_t)(node0 + row) * D_ + k]
                           : ab[(size_t)(node0 + row) * D_ + (k - D_)];
        catl[row][k] = (_Float16)v;
    }
    __syncthreads();

    int wave  = tid >> 5;            // output-column tile (o in [16w, 16w+16))
    int lane  = tid & 31;
    int mrow  = lane & 15;
    int khalf = lane >> 4;           // 0: lanes 0-15, 1: lanes 16-31
    int ocol  = wave * 16 + (lane & 15);

    v8f c = {0.f, 0.f, 0.f, 0.f, 0.f, 0.f, 0.f, 0.f};
    for (int kk = 0; kk < 2 * D_; kk += 32) {
        v16h a, bf;
        // A fragment: a[i] = cat[mrow, ka+i] (16 contiguous k, same order as B)
        int ka = kk + khalf * 16;
#pragma unroll
        for (int i = 0; i < 8; ++i) {
            a[i]     = catl[mrow][ka + i];
            a[8 + i] = catl[mrow][ka + 8 + i];
        }
        // B fragment: b[i] = Wt[kb+i, ocol] = lin_w[ocol, kb+i] (16 contiguous halves)
        int kb = kk + khalf * 16;
        const _Float16* wr = w16 + (size_t)ocol * (2 * D_) + kb;
#pragma unroll
        for (int i = 0; i < 16; ++i) bf[i] = wr[i];

        c = __builtin_amdgcn_wmma_f32_16x16x32_f16(false, a, false, bf,
                                                   (short)0, c, false, false);
    }

    // spill C tile to LDS with bias (C layout: lane<16 -> rows 0-7, else rows 8-15)
    int crow0 = khalf * 8;
#pragma unroll
    for (int v = 0; v < 8; ++v) hidl[crow0 + v][ocol] = c[v] + bias[ocol];
    __syncthreads();

    // LayerNorm over D=64 per node row: 8 threads/row partials, then reduce
    int row = tid >> 3, sub = tid & 7;
    float s = 0.f, ss = 0.f;
#pragma unroll
    for (int i = 0; i < 8; ++i) {
        float h = hidl[row][sub * 8 + i];
        s += h; ss += h * h;
    }
    red[row][sub][0] = s; red[row][sub][1] = ss;
    __syncthreads();
    if (sub == 0) {
        float S = 0.f, SS = 0.f;
        for (int i = 0; i < 8; ++i) { S += red[row][i][0]; SS += red[row][i][1]; }
        float m = S / (float)D_;
        red[row][0][0] = m;
        red[row][0][1] = rsqrtf(SS / (float)D_ - m * m + EPS_);
    }
    __syncthreads();
    float mn  = red[row][0][0];
    float inv = red[row][0][1];
    float* xout = x + ((size_t)b * N_) * D_;
#pragma unroll
    for (int i = 0; i < 8; ++i) {
        int o = sub * 8 + i;
        float h = (hidl[row][o] - mn) * inv * lns[o] + lnb[o];
        h = fmaxf(h, 0.f);
        size_t gi = (size_t)(node0 + row) * D_ + o;
        xout[gi] = h + xb[gi];
    }
}

// ----------------------------------------------------------------------------
// tokens[(b*NNEG+j), w, d] = x[b, tsel[b,j], d] + pe[w,d]
// ----------------------------------------------------------------------------
__global__ void wt_gather_tokens_kernel(const float* __restrict__ x,
                                        const int* __restrict__ tsel,
                                        const float* __restrict__ pe,
                                        float* __restrict__ tokens, int w)
{
    int idx = blockIdx.x * blockDim.x + threadIdx.x;
    if (idx >= B_ * NNEG_ * D_) return;
    int d  = idx & 63;
    int bj = idx >> 6;             // b*NNEG + j
    int b  = bj / NNEG_;
    int t  = tsel[bj];
    tokens[(size_t)bj * W_ * D_ + (size_t)w * D_ + d] =
        x[((size_t)b * N_ + t) * D_ + d] + pe[w * D_ + d];
}

// ----------------------------------------------------------------------------
// 2-layer transformer over (W=3, D=64) sequences + final scoring MLP.
// one block (64 threads) per sequence; everything in LDS.
// ----------------------------------------------------------------------------
__global__ void wt_trans_score_kernel(const float* __restrict__ tokens,
                                      const float* __restrict__ query,
                                      const float* __restrict__ wq,
                                      const float* __restrict__ wk,
                                      const float* __restrict__ wv,
                                      const float* __restrict__ wo,
                                      const float* __restrict__ ffw1,
                                      const float* __restrict__ ffb1,
                                      const float* __restrict__ ffw2,
                                      const float* __restrict__ ffb2,
                                      const float* __restrict__ f1s,
                                      const float* __restrict__ f1b,
                                      const float* __restrict__ f2s,
                                      const float* __restrict__ f2b,
                                      const float* __restrict__ m1w,
                                      const float* __restrict__ m1b,
                                      const float* __restrict__ m2w,
                                      const float* __restrict__ m2b,
                                      float* __restrict__ out)
{
    __shared__ float tok[W_][D_];
    __shared__ float qv[W_][D_], kv[W_][D_], vv[W_][D_];
    __shared__ float att[4][W_][W_];
    __shared__ float tmp[W_][D_];
    __shared__ float stats[W_][2];
    __shared__ float ff1[W_][FFN_];
    __shared__ float hidb[2 * D_];
    __shared__ float redb[D_];

    int seq = blockIdx.x;            // b*NNEG + j
    int b   = seq / NNEG_;
    int d   = threadIdx.x;           // 0..63

    for (int t = 0; t < W_; ++t) tok[t][d] = tokens[((size_t)seq * W_ + t) * D_ + d];
    __syncthreads();

    for (int f = 0; f < FLAYERS_; ++f) {
        const float* WQ = wq + (size_t)f * D_ * D_;
        const float* WK = wk + (size_t)f * D_ * D_;
        const float* WV = wv + (size_t)f * D_ * D_;
        const float* WO = wo + (size_t)f * D_ * D_;
        const float* W1 = ffw1 + (size_t)f * FFN_ * D_;
        const float* B1 = ffb1 + (size_t)f * FFN_;
        const float* W2 = ffw2 + (size_t)f * D_ * FFN_;
        const float* B2 = ffb2 + (size_t)f * D_;
        const float* S1 = f1s + (size_t)f * D_; const float* Bb1 = f1b + (size_t)f * D_;
        const float* S2 = f2s + (size_t)f * D_; const float* Bb2 = f2b + (size_t)f * D_;

        // q,k,v projections (row d of each weight is contiguous)
        for (int t = 0; t < W_; ++t) {
            float aq = 0.f, ak = 0.f, av = 0.f;
#pragma unroll 8
            for (int k = 0; k < D_; ++k) {
                float tv = tok[t][k];
                aq += tv * WQ[d * D_ + k];
                ak += tv * WK[d * D_ + k];
                av += tv * WV[d * D_ + k];
            }
            qv[t][d] = aq; kv[t][d] = ak; vv[t][d] = av;
        }
        __syncthreads();

        if (d < 36) {                 // 4 heads x 3x3 score matrix
            int h = d / 9, rem = d % 9, tq = rem / 3, tk = rem % 3;
            float sc = 0.f;
#pragma unroll
            for (int i = 0; i < 16; ++i) sc += qv[tq][h * 16 + i] * kv[tk][h * 16 + i];
            att[h][tq][tk] = sc * 0.25f;   // 1/sqrt(hd=16)
        }
        __syncthreads();
        if (d < 12) {                 // softmax over tk
            int h = d / 3, tq = d % 3;
            float a0 = att[h][tq][0], a1 = att[h][tq][1], a2 = att[h][tq][2];
            float mx = fmaxf(a0, fmaxf(a1, a2));
            float e0 = expf(a0 - mx), e1 = expf(a1 - mx), e2 = expf(a2 - mx);
            float in = 1.f / (e0 + e1 + e2);
            att[h][tq][0] = e0 * in; att[h][tq][1] = e1 * in; att[h][tq][2] = e2 * in;
        }
        __syncthreads();
        {
            int h = d >> 4;
            for (int t = 0; t < W_; ++t)
                tmp[t][d] = att[h][t][0] * vv[0][d] + att[h][t][1] * vv[1][d]
                          + att[h][t][2] * vv[2][d];
        }
        __syncthreads();
        // out projection + residual -> qv (reused as pre-LN y)
        for (int t = 0; t < W_; ++t) {
            float acc = 0.f;
#pragma unroll 8
            for (int k = 0; k < D_; ++k) acc += tmp[t][k] * WO[d * D_ + k];
            qv[t][d] = tok[t][d] + acc;
        }
        __syncthreads();
        if (d < W_) {
            float s = 0.f, ss = 0.f;
            for (int k = 0; k < D_; ++k) { float v = qv[d][k]; s += v; ss += v * v; }
            float m = s / (float)D_;
            stats[d][0] = m; stats[d][1] = rsqrtf(ss / (float)D_ - m * m + EPS_);
        }
        __syncthreads();
        for (int t = 0; t < W_; ++t)
            tok[t][d] = (qv[t][d] - stats[t][0]) * stats[t][1] * S1[d] + Bb1[d];
        __syncthreads();

        // FFN: 64 -> 256 (relu) -> 64
        for (int jj = 0; jj < FFN_ / D_; ++jj) {
            int j = jj * D_ + d;
            const float* w1r = W1 + (size_t)j * D_;
            for (int t = 0; t < W_; ++t) {
                float acc = B1[j];
#pragma unroll 8
                for (int k = 0; k < D_; ++k) acc += tok[t][k] * w1r[k];
                ff1[t][j] = fmaxf(acc, 0.f);
            }
        }
        __syncthreads();
        for (int t = 0; t < W_; ++t) {
            float acc = B2[d];
            const float* w2r = W2 + (size_t)d * FFN_;
#pragma unroll 8
            for (int k = 0; k < FFN_; ++k) acc += ff1[t][k] * w2r[k];
            qv[t][d] = tok[t][d] + acc;
        }
        __syncthreads();
        if (d < W_) {
            float s = 0.f, ss = 0.f;
            for (int k = 0; k < D_; ++k) { float v = qv[d][k]; s += v; ss += v * v; }
            float m = s / (float)D_;
            stats[d][0] = m; stats[d][1] = rsqrtf(ss / (float)D_ - m * m + EPS_);
        }
        __syncthreads();
        for (int t = 0; t < W_; ++t)
            tok[t][d] = (qv[t][d] - stats[t][0]) * stats[t][1] * S2[d] + Bb2[d];
        __syncthreads();
    }

    // scoring MLP on feat = [tok[W-1] | query[b]]  (128 -> 128 relu -> 1)
    const float* qb = query + (size_t)b * D_;
    for (int ii = 0; ii < 2; ++ii) {
        int i = ii * D_ + d;
        const float* wr = m1w + (size_t)i * 2 * D_;
        float acc = m1b[i];
#pragma unroll 8
        for (int k = 0; k < D_; ++k) acc += tok[W_ - 1][k] * wr[k];
#pragma unroll 8
        for (int k = 0; k < D_; ++k) acc += qb[k] * wr[D_ + k];
        hidb[i] = fmaxf(acc, 0.f);
    }
    __syncthreads();
    redb[d] = hidb[d] * m2w[d] + hidb[d + D_] * m2w[d + D_];
    __syncthreads();
    for (int s = 32; s > 0; s >>= 1) {
        if (d < s) redb[d] += redb[d + s];
        __syncthreads();
    }
    if (d == 0) out[seq] = redb[0] + m2b[0];
}

// ----------------------------------------------------------------------------
extern "C" void kernel_launch(void* const* d_in, const int* in_sizes, int n_in,
                              void* d_out, int out_size, void* d_ws, size_t ws_size,
                              hipStream_t stream)
{
    const int*   qt   = (const int*)  d_in[0];
    const int*   eidx = (const int*)  d_in[1];
    const int*   etyp = (const int*)  d_in[2];
    const int*   etim = (const int*)  d_in[3];
    const float* qemb = (const float*)d_in[4];
    const float* ew   = (const float*)d_in[5];
    const float* rlw  = (const float*)d_in[6];
    const float* rlb  = (const float*)d_in[7];
    const float* tw   = (const float*)d_in[8];
    const float* tb   = (const float*)d_in[9];
    const float* linw = (const float*)d_in[10];
    const float* linb = (const float*)d_in[11];
    const float* lns  = (const float*)d_in[12];
    const float* lnb  = (const float*)d_in[13];
    const float* pe   = (const float*)d_in[14];
    const float* wq   = (const float*)d_in[15];
    const float* wk   = (const float*)d_in[16];
    const float* wv   = (const float*)d_in[17];
    const float* wo   = (const float*)d_in[18];
    const float* ffw1 = (const float*)d_in[19];
    const float* ffb1 = (const float*)d_in[20];
    const float* ffw2 = (const float*)d_in[21];
    const float* ffb2 = (const float*)d_in[22];
    const float* f1s  = (const float*)d_in[23];
    const float* f1b  = (const float*)d_in[24];
    const float* f2s  = (const float*)d_in[25];
    const float* f2b  = (const float*)d_in[26];
    const float* m1w  = (const float*)d_in[27];
    const float* m1b  = (const float*)d_in[28];
    const float* m2w  = (const float*)d_in[29];
    const float* m2b  = (const float*)d_in[30];

    // workspace carve-out (all fits in ~21.3 MB; L2-resident on MI455X)
    char* ws = (char*)d_ws;
    size_t off = 0;
    auto carve = [&](size_t bytes) {
        size_t o = off;
        off = (off + bytes + 255) & ~(size_t)255;
        return o;
    };
    float*    x      = (float*)   (ws + carve(sizeof(float) * B_ * N_ * D_));
    float*    agg    = (float*)   (ws + carve(sizeof(float) * B_ * N_ * D_));
    float*    relall = (float*)   (ws + carve(sizeof(float) * L_ * B_ * R_ * D_));
    float*    tokens = (float*)   (ws + carve(sizeof(float) * B_ * NNEG_ * W_ * D_));
    float*    query  = (float*)   (ws + carve(sizeof(float) * B_ * D_));
    _Float16* w16    = (_Float16*)(ws + carve(sizeof(_Float16) * L_ * D_ * 2 * D_));
    int*      h0     = (int*)     (ws + carve(sizeof(int) * B_));
    int*      tsel   = (int*)     (ws + carve(sizeof(int) * B_ * NNEG_));
    int*      mint   = (int*)     (ws + carve(sizeof(int)));
    (void)ws_size; (void)in_sizes; (void)n_in; (void)out_size;

    wt_prep_kernel<<<1, 256, 0, stream>>>(qt, qemb, etim, h0, tsel, query, mint);
    wt_convw16_kernel<<<(L_ * D_ * 2 * D_ + 255) / 256, 256, 0, stream>>>(linw, w16);
    wt_rel_kernel<<<(L_ * B_ * R_ * D_ + 255) / 256, 256, 0, stream>>>(query, rlw, rlb, relall);

    const int initBlocks = (B_ * N_ * D_ + 255) / 256;
    for (int w = 0; w < W_; ++w) {
        wt_init_state_kernel<<<initBlocks, 256, 0, stream>>>(x, query, h0);
        for (int l = 0; l < L_; ++l) {
            wt_init_state_kernel<<<initBlocks, 256, 0, stream>>>(agg, query, h0);
            wt_edge_kernel<<<(E_ * 32 + 255) / 256, 256, 0, stream>>>(
                eidx + (size_t)(w * 2 + 0) * E_,
                eidx + (size_t)(w * 2 + 1) * E_,
                etyp + (size_t)w * E_,
                etim + (size_t)w * E_,
                ew   + (size_t)w * E_,
                x,
                relall + (size_t)l * B_ * R_ * D_,
                tw + (size_t)l * R_ * D_,
                tb + (size_t)l * R_ * D_,
                mint, agg);
            wt_gemm_ln_kernel<<<B_ * (N_ / 16), 128, 0, stream>>>(
                x, agg,
                w16 + (size_t)l * D_ * 2 * D_,
                linb + (size_t)l * D_,
                lns  + (size_t)l * D_,
                lnb  + (size_t)l * D_);
        }
        wt_gather_tokens_kernel<<<(B_ * NNEG_ * D_ + 255) / 256, 256, 0, stream>>>(
            x, tsel, pe, tokens, w);
    }

    wt_trans_score_kernel<<<B_ * NNEG_, 64, 0, stream>>>(
        tokens, query, wq, wk, wv, wo, ffw1, ffb1, ffw2, ffb2,
        f1s, f1b, f2s, f2b, m1w, m1b, m2w, m2b, (float*)d_out);
}
